// PretrainedFeatureExtractor_25074019074102
// MI455X (gfx1250) — compile-verified, matrix operand order checked by reading, not run on hardware
//
#include <hip/hip_runtime.h>

typedef __attribute__((ext_vector_type(16))) __bf16 v16bf;
typedef __attribute__((ext_vector_type(8)))  __bf16 v8bf;
typedef __attribute__((ext_vector_type(8)))  float  v8f;

#define WMMA_BF16(a, b, c) \
  __builtin_amdgcn_wmma_f32_16x16x32_bf16(false, (a), false, (b), (short)0, (c), false, false)

// ---------------------------------------------------------------------------
// Load one 16x32 bf16 fragment (A or B layout; symmetric) from a pre-converted
// row-major bf16 matrix. Per the CDNA5 ISA 16-bit fragment layout, lane
// (m = lane&15, khalf = lane>>4) holds K = khalf*8 + [0..7] (elements 0..7)
// and K = 16 + khalf*8 + [0..7] (elements 8..15): two aligned 16-byte chunks.
// ---------------------------------------------------------------------------
__device__ __forceinline__ v16bf load_frag_bf(const __bf16* __restrict__ rowbase,
                                              int k0, int khalf) {
  const v8bf* p = (const v8bf*)(rowbase + k0 + khalf * 8);
  return __builtin_shufflevector(p[0], p[2],
                                 0, 1, 2, 3, 4, 5, 6, 7, 8, 9, 10, 11, 12, 13, 14, 15);
}

// f32 -> bf16 hi/lo split (Ootomo bf16x3 scheme), done once per matrix.
__global__ __launch_bounds__(256)
void cvt_hilo_kernel(const float* __restrict__ x, __bf16* __restrict__ hi,
                     __bf16* __restrict__ lo, int n) {
  int i = blockIdx.x * 256 + threadIdx.x;
  if (i >= n) return;
  float  v = x[i];
  __bf16 h = (__bf16)v;
  hi[i] = h;
  lo[i] = (__bf16)(v - (float)h);
}

// ---------------------------------------------------------------------------
// GEMM: C[M,N] = A[M,K] @ W[N,K]^T + bias, bf16x3 WMMA, f32 accumulate.
// One wave per 16x32 output tile (A fragment reused across two B tiles).
// grid = (N/32, M/64), block = 128 (4 waves). K must be a multiple of 32.
// ---------------------------------------------------------------------------
__global__ __launch_bounds__(128)
void gemm_bf16x3_kernel(const __bf16* __restrict__ Ahi, const __bf16* __restrict__ Alo, int lda,
                        const __bf16* __restrict__ Whi, const __bf16* __restrict__ Wlo,
                        const float* __restrict__ bias,
                        float* __restrict__ C, int ldc, int coff, int K) {
  int tid   = threadIdx.x;
  int wave  = tid >> 5;
  int lane  = tid & 31;
  int mTile = blockIdx.y * 4 + wave;
  int nPair = blockIdx.x;                 // 32 output columns
  int m     = lane & 15;
  int khalf = lane >> 4;

  const __bf16* ah  = Ahi + (size_t)(mTile * 16 + m) * lda;
  const __bf16* al  = Alo + (size_t)(mTile * 16 + m) * lda;
  const __bf16* w0h = Whi + (size_t)(nPair * 32 + m) * K;
  const __bf16* w0l = Wlo + (size_t)(nPair * 32 + m) * K;
  const __bf16* w1h = w0h + (size_t)16 * K;
  const __bf16* w1l = w0l + (size_t)16 * K;

  v8f acc0 = {}, acc1 = {};
  for (int k0 = 0; k0 < K; k0 += 32) {
    __builtin_prefetch(ah + k0 + 128, 0, 0);
    __builtin_prefetch(w0h + k0 + 128, 0, 0);
    v16bf fah  = load_frag_bf(ah,  k0, khalf);
    v16bf fal  = load_frag_bf(al,  k0, khalf);
    v16bf fb0h = load_frag_bf(w0h, k0, khalf);
    v16bf fb0l = load_frag_bf(w0l, k0, khalf);
    v16bf fb1h = load_frag_bf(w1h, k0, khalf);
    v16bf fb1l = load_frag_bf(w1l, k0, khalf);
    acc0 = WMMA_BF16(fah, fb0h, acc0);
    acc0 = WMMA_BF16(fah, fb0l, acc0);
    acc0 = WMMA_BF16(fal, fb0h, acc0);
    acc1 = WMMA_BF16(fah, fb1h, acc1);
    acc1 = WMMA_BF16(fah, fb1l, acc1);
    acc1 = WMMA_BF16(fal, fb1h, acc1);
  }

  int   n0    = nPair * 32 + m;
  float bv0   = bias[n0];
  float bv1   = bias[n0 + 16];
  int   rbase = mTile * 16 + (khalf << 3);
#pragma unroll
  for (int i = 0; i < 8; ++i) {
    C[(size_t)(rbase + i) * ldc + coff + n0]      = acc0[i] + bv0;
    C[(size_t)(rbase + i) * ldc + coff + n0 + 16] = acc1[i] + bv1;
  }
}

// Column mean / rsqrt(var+eps) over `rows` rows. One block per column.
__global__ __launch_bounds__(256)
void col_stats_kernel(const float* __restrict__ X, int ldx, int coff, int rows,
                      float* __restrict__ mean, float* __restrict__ rstd) {
  int c = blockIdx.x;
  int t = threadIdx.x;
  float s = 0.f, s2 = 0.f;
  for (int r = t; r < rows; r += 256) {
    float v = X[(size_t)r * ldx + coff + c];
    s += v; s2 += v * v;
  }
  __shared__ float sh[256], sh2[256];
  sh[t] = s; sh2[t] = s2;
  __syncthreads();
  for (int o = 128; o > 0; o >>= 1) {
    if (t < o) { sh[t] += sh[t + o]; sh2[t] += sh2[t + o]; }
    __syncthreads();
  }
  if (t == 0) {
    float m = sh[0] / (float)rows;
    float v = sh2[0] / (float)rows - m * m;
    mean[c] = m;
    rstd[c] = rsqrtf(v + 1e-5f);
  }
}

__global__ void bn_relu_kernel(const float* __restrict__ X, int ldx, int xoff,
                               float* __restrict__ Y, int ldy, int yoff,
                               const float* __restrict__ mean, const float* __restrict__ rstd,
                               const float* __restrict__ g, const float* __restrict__ b,
                               int total, int cols) {
  int idx = blockIdx.x * blockDim.x + threadIdx.x;
  if (idx >= total) return;
  int r = idx / cols, c = idx - r * cols;
  float v = (X[(size_t)r * ldx + xoff + c] - mean[c]) * rstd[c] * g[c] + b[c];
  Y[(size_t)r * ldy + yoff + c] = v > 0.f ? v : 0.f;
}

// lin1: 3 -> 64 (K too small for WMMA)
__global__ void lin1_kernel(const float* __restrict__ pc, const float* __restrict__ w,
                            const float* __restrict__ b, float* __restrict__ h1) {
  int idx = blockIdx.x * blockDim.x + threadIdx.x;
  int r = idx >> 6, c = idx & 63;
  const float* p = pc + (size_t)r * 3;
  h1[idx] = p[0] * w[c * 3] + p[1] * w[c * 3 + 1] + p[2] * w[c * 3 + 2] + b[c];
}

__global__ __launch_bounds__(256)
void rowsq_kernel(const float* __restrict__ comb, float* __restrict__ sq, int rows) {
  int r = blockIdx.x * blockDim.x + threadIdx.x;
  if (r >= rows) return;
  const float* p = comb + (size_t)r * 256;
  float s = 0.f;
#pragma unroll 8
  for (int d = 0; d < 128; ++d) s += p[d] * p[d];
  sq[r] = s;
}

// Sorted-ascending top-16 insertion; fully unrolled ladder keeps arrays in VGPRs.
__device__ __forceinline__ void topk_insert(float (&v)[16], int (&ix)[16], float s, int id) {
  if (s < v[15]) {
    v[15] = s; ix[15] = id;
#pragma unroll
    for (int j = 15; j > 0; --j) {
      if (v[j] < v[j - 1]) {
        float tv = v[j]; v[j] = v[j - 1]; v[j - 1] = tv;
        int   ti = ix[j]; ix[j] = ix[j - 1]; ix[j - 1] = ti;
      }
    }
  }
}

// ---------------------------------------------------------------------------
// KNN + max-pooled edge features. One block = 16 query rows of one batch.
// WMMA (bf16x3, pre-converted fragments) computes 16x16 tiles of
// score = ||x_m||^2 - 2 x_n.x_m (||x_n||^2 is rank-invariant per query row),
// staged through LDS; per-thread register top-16 lists over 1/8 of columns;
// 8-way merge in LDS; then gather+max of neighbor features.
// grid = (128, 8), block = 128.
// ---------------------------------------------------------------------------
__global__ __launch_bounds__(128)
void knn_edge_kernel(const __bf16* __restrict__ chi, const __bf16* __restrict__ clo,
                     const float* __restrict__ comb, const float* __restrict__ sq,
                     float* __restrict__ out) {
  const int NP = 2048, LD = 256;
  int b   = blockIdx.y;
  int rt  = blockIdx.x;                // 16-row tile within batch
  int tid = threadIdx.x;
  int wave = tid >> 5, lane = tid & 31;
  int m = lane & 15, khalf = lane >> 4;
  size_t base = (size_t)b * NP;

  __shared__ float s_sc[16 * 256];
  __shared__ float s_cv[16 * 128];
  __shared__ int   s_ci[16 * 128];
  __shared__ int   s_nbr[16 * 16];

  float tv[16]; int tix[16];
#pragma unroll
  for (int j = 0; j < 16; ++j) { tv[j] = 3.0e38f; tix[j] = 0; }

  int r16  = tid & 15;   // scan row
  int part = tid >> 4;   // 0..7 column segment

  // Query-side fragments reused across all 128 column tiles: load once.
  const __bf16* arh = chi + (base + rt * 16 + m) * LD;
  const __bf16* arl = clo + (base + rt * 16 + m) * LD;
  v16bf ahi[4], alo[4];
#pragma unroll
  for (int k = 0; k < 4; ++k) {
    ahi[k] = load_frag_bf(arh, k * 32, khalf);
    alo[k] = load_frag_bf(arl, k * 32, khalf);
  }

  for (int chunk = 0; chunk < NP; chunk += 256) {
#pragma unroll 1
    for (int sub = 0; sub < 4; ++sub) {
      int colTile = chunk + (wave * 4 + sub) * 16;
      const __bf16* brh = chi + (base + colTile + m) * LD;
      const __bf16* brl = clo + (base + colTile + m) * LD;
      v8f acc = {};
#pragma unroll
      for (int k = 0; k < 4; ++k) {
        v16bf bh = load_frag_bf(brh, k * 32, khalf);
        v16bf bl = load_frag_bf(brl, k * 32, khalf);
        acc = WMMA_BF16(ahi[k], bh, acc);
        acc = WMMA_BF16(ahi[k], bl, acc);
        acc = WMMA_BF16(alo[k], bh, acc);
      }
      float sqc   = sq[base + colTile + m];
      int   mbase = khalf << 3;
#pragma unroll
      for (int i = 0; i < 8; ++i)
        s_sc[(mbase + i) * 256 + (colTile - chunk) + m] = sqc - 2.0f * acc[i];
    }
    __syncthreads();
    int cbase = part * 32;
#pragma unroll 4
    for (int j = 0; j < 32; ++j) {
      int col = cbase + j;
      topk_insert(tv, tix, s_sc[r16 * 256 + col], chunk + col);
    }
    __syncthreads();
  }

#pragma unroll
  for (int j = 0; j < 16; ++j) {
    s_cv[r16 * 128 + part * 16 + j] = tv[j];
    s_ci[r16 * 128 + part * 16 + j] = tix[j];
  }
  __syncthreads();
  if (tid < 16) {
    float fv[16]; int fix[16];
#pragma unroll
    for (int j = 0; j < 16; ++j) { fv[j] = 3.0e38f; fix[j] = 0; }
    for (int j = 0; j < 128; ++j)
      topk_insert(fv, fix, s_cv[tid * 128 + j], s_ci[tid * 128 + j]);
#pragma unroll
    for (int j = 0; j < 16; ++j) s_nbr[tid * 16 + j] = fix[j];
  }
  __syncthreads();

  int gr = tid >> 3, seg = tid & 7;
  size_t grow = base + rt * 16 + gr;
#pragma unroll 1
  for (int d = seg * 16; d < seg * 16 + 16; ++d) {
    float mx = -3.0e38f;
#pragma unroll
    for (int j = 0; j < 16; ++j) {
      int nb = s_nbr[gr * 16 + j];
      mx = fmaxf(mx, comb[(base + nb) * LD + d]);
    }
    out[grow * LD + 128 + d] = mx;
  }
}

// Per-batch max over 2048 points of enh (cols 0..127 of cg). grid=8, block=128.
__global__ __launch_bounds__(128)
void gin_max_kernel(const float* __restrict__ cg, float* __restrict__ gin) {
  int b = blockIdx.x, c = threadIdx.x;
  float mx = -3.0e38f;
  for (int r = 0; r < 2048; ++r)
    mx = fmaxf(mx, cg[((size_t)b * 2048 + r) * 640 + c]);
  gin[b * 128 + c] = mx;
}

// Tiny head MLP: [8,128] -> 256 -> 512 with BN over the 8-row batch. 1 block.
__global__ __launch_bounds__(512)
void head_kernel(const float* __restrict__ gin,
                 const float* __restrict__ g1w, const float* __restrict__ g1b,
                 const float* __restrict__ gbn1g, const float* __restrict__ gbn1b,
                 const float* __restrict__ g2w, const float* __restrict__ g2b,
                 const float* __restrict__ gbn2g, const float* __restrict__ gbn2b,
                 float* __restrict__ glob) {
  __shared__ float s_in[8 * 128];
  __shared__ float s_h[8 * 256];
  __shared__ float s_o[8 * 512];
  int t = threadIdx.x;
  for (int i = t; i < 8 * 128; i += 512) s_in[i] = gin[i];
  __syncthreads();
  for (int i = t; i < 8 * 256; i += 512) {
    int r = i >> 8, c = i & 255;
    float s = g1b[c];
    for (int k = 0; k < 128; ++k) s += s_in[r * 128 + k] * g1w[c * 128 + k];
    s_h[i] = s;
  }
  __syncthreads();
  if (t < 256) {
    float mean = 0.f;
    for (int r = 0; r < 8; ++r) mean += s_h[r * 256 + t];
    mean *= 0.125f;
    float var = 0.f;
    for (int r = 0; r < 8; ++r) { float d = s_h[r * 256 + t] - mean; var += d * d; }
    var *= 0.125f;
    float rs = rsqrtf(var + 1e-5f);
    for (int r = 0; r < 8; ++r) {
      float x = (s_h[r * 256 + t] - mean) * rs * gbn1g[t] + gbn1b[t];
      s_h[r * 256 + t] = x > 0.f ? x : 0.f;
    }
  }
  __syncthreads();
  for (int i = t; i < 8 * 512; i += 512) {
    int r = i >> 9, c = i & 511;
    float s = g2b[c];
    for (int k = 0; k < 256; ++k) s += s_h[r * 256 + k] * g2w[c * 256 + k];
    s_o[i] = s;
  }
  __syncthreads();
  {
    float mean = 0.f;
    for (int r = 0; r < 8; ++r) mean += s_o[r * 512 + t];
    mean *= 0.125f;
    float var = 0.f;
    for (int r = 0; r < 8; ++r) { float d = s_o[r * 512 + t] - mean; var += d * d; }
    var *= 0.125f;
    float rs = rsqrtf(var + 1e-5f);
    for (int r = 0; r < 8; ++r) {
      float x = (s_o[r * 512 + t] - mean) * rs * gbn2g[t] + gbn2b[t];
      glob[r * 512 + t] = x > 0.f ? x : 0.f;
    }
  }
}

__global__ void bcast_glob_kernel(const float* __restrict__ glob, float* __restrict__ cg) {
  int idx = blockIdx.x * blockDim.x + threadIdx.x;   // over 16384*512
  int r = idx >> 9, c = idx & 511;
  cg[(size_t)r * 640 + 128 + c] = glob[(r >> 11) * 512 + c];
}

__global__ void final_kernel(const float* __restrict__ po,
                             const float* __restrict__ mean, const float* __restrict__ rstd,
                             const float* __restrict__ g, const float* __restrict__ b,
                             const float* __restrict__ cat_bias, const int* __restrict__ cat_ids,
                             float* __restrict__ out) {
  int idx = blockIdx.x * blockDim.x + threadIdx.x;   // over 16384*512
  int r = idx >> 9, c = idx & 511;
  float v = (po[idx] - mean[c]) * rstd[c] * g[c] + b[c];
  v = v > 0.f ? v : 0.f;
  out[idx] = v + 0.1f * cat_bias[cat_ids[r >> 11] * 512 + c];
}

// ---------------------------------------------------------------------------
extern "C" void kernel_launch(void* const* d_in, const int* in_sizes, int n_in,
                              void* d_out, int out_size, void* d_ws, size_t ws_size,
                              hipStream_t stream) {
  const int R = 16384;                       // B*N = 8*2048
  const float* pc    = (const float*)d_in[0];
  const float* l1w   = (const float*)d_in[1];
  const float* l1b   = (const float*)d_in[2];
  const float* bn1g  = (const float*)d_in[3];
  const float* bn1b  = (const float*)d_in[4];
  const float* l2w   = (const float*)d_in[5];
  const float* l2b   = (const float*)d_in[6];
  const float* bn2g  = (const float*)d_in[7];
  const float* bn2b  = (const float*)d_in[8];
  const float* e1w   = (const float*)d_in[9];
  const float* e1b   = (const float*)d_in[10];
  const float* ebn1g = (const float*)d_in[11];
  const float* ebn1b = (const float*)d_in[12];
  const float* e2w   = (const float*)d_in[13];
  const float* e2b   = (const float*)d_in[14];
  const float* ebn2g = (const float*)d_in[15];
  const float* ebn2b = (const float*)d_in[16];
  const float* g1w   = (const float*)d_in[17];
  const float* g1b   = (const float*)d_in[18];
  const float* gbn1g = (const float*)d_in[19];
  const float* gbn1b = (const float*)d_in[20];
  const float* g2w   = (const float*)d_in[21];
  const float* g2b   = (const float*)d_in[22];
  const float* gbn2g = (const float*)d_in[23];
  const float* gbn2b = (const float*)d_in[24];
  const float* pw    = (const float*)d_in[25];
  const float* pb    = (const float*)d_in[26];
  const float* pbng  = (const float*)d_in[27];
  const float* pbnb  = (const float*)d_in[28];
  const float* catb  = (const float*)d_in[29];
  const int*   cats  = (const int*)d_in[30];
  float* out = (float*)d_out;

  // ---- workspace layout (bytes), with lifetime-based aliasing ----
  char* Wb = (char*)d_ws;
  float* h1   = (float*)(Wb + 0);            //  4,194,304  B  [R x  64] f32
  float* comb = (float*)(Wb + 4194304);      // 16,777,216  B  [R x 256] f32
  float* e1o  = (float*)(Wb + 20971520);     //  8,388,608  B  [R x 128] f32
  float* e2o  = (float*)(Wb + 29360128);     //  8,388,608  B  [R x 128] f32
  float* sq   = (float*)(Wb + 37748736);     //     65,536  B  [R] f32
  float* po   = (float*)(Wb + 0);            // 33,554,432  B  alias h1..e2o (dead by then)
  float* cg   = (float*)(Wb + 37814272);     // 41,943,040  B  [R x 640] f32
  __bf16* slab = (__bf16*)(Wb + 79757312);   // 41,943,040  B  shared bf16 hi+lo slab
  __bf16* l2wbf = (__bf16*)(Wb + 121700352); //     32,768  B
  __bf16* e1wbf = (__bf16*)(Wb + 121733120); //    131,072  B
  __bf16* e2wbf = (__bf16*)(Wb + 121864192); //     65,536  B
  __bf16* pwbf  = (__bf16*)(Wb + 121929728); //  1,310,720  B
  float* gin  = (float*)(Wb + 123240448);    //      4,096  B
  float* glob = (float*)(Wb + 123244544);    //     16,384  B
  float* mb   = (float*)(Wb + 123260928);    //      2,560  B
  float* rb   = (float*)(Wb + 123263488);    //      2,560  B

  // weight conversions (once)
  cvt_hilo_kernel<<<(128 * 64 + 255) / 256, 256, 0, stream>>>(l2w, l2wbf, l2wbf + 128 * 64, 128 * 64);
  cvt_hilo_kernel<<<(128 * 256 + 255) / 256, 256, 0, stream>>>(e1w, e1wbf, e1wbf + 128 * 256, 128 * 256);
  cvt_hilo_kernel<<<(128 * 128 + 255) / 256, 256, 0, stream>>>(e2w, e2wbf, e2wbf + 128 * 128, 128 * 128);
  cvt_hilo_kernel<<<(512 * 640 + 255) / 256, 256, 0, stream>>>(pw, pwbf, pwbf + 512 * 640, 512 * 640);

  // local encoder: 3 -> 64 -> 128
  lin1_kernel<<<R * 64 / 256, 256, 0, stream>>>(pc, l1w, l1b, h1);
  col_stats_kernel<<<64, 256, 0, stream>>>(h1, 64, 0, R, mb, rb);
  bn_relu_kernel<<<R * 64 / 256, 256, 0, stream>>>(h1, 64, 0, h1, 64, 0, mb, rb, bn1g, bn1b, R * 64, 64);
  cvt_hilo_kernel<<<R * 64 / 256, 256, 0, stream>>>(h1, slab, slab + R * 64, R * 64);
  gemm_bf16x3_kernel<<<dim3(4, R / 64), 128, 0, stream>>>(slab, slab + R * 64, 64,
                                                          l2wbf, l2wbf + 128 * 64, l2b,
                                                          comb, 256, 0, 64);
  col_stats_kernel<<<128, 256, 0, stream>>>(comb, 256, 0, R, mb, rb);
  bn_relu_kernel<<<R * 128 / 256, 256, 0, stream>>>(comb, 256, 0, comb, 256, 0, mb, rb, bn2g, bn2b, R * 128, 128);

  // knn max-pooled edge features -> comb cols 128..255
  cvt_hilo_kernel<<<R * 256 / 256, 256, 0, stream>>>(comb, slab, slab + R * 256, R * 256);
  rowsq_kernel<<<R / 256, 256, 0, stream>>>(comb, sq, R);
  knn_edge_kernel<<<dim3(128, 8), 128, 0, stream>>>(slab, slab + R * 256, comb, sq, comb);

  // edge encoder: 256 -> 128 -> 128 (enh written into cg cols 0..127)
  cvt_hilo_kernel<<<R * 256 / 256, 256, 0, stream>>>(comb, slab, slab + R * 256, R * 256);
  gemm_bf16x3_kernel<<<dim3(4, R / 64), 128, 0, stream>>>(slab, slab + R * 256, 256,
                                                          e1wbf, e1wbf + 128 * 256, e1b,
                                                          e1o, 128, 0, 256);
  col_stats_kernel<<<128, 256, 0, stream>>>(e1o, 128, 0, R, mb, rb);
  bn_relu_kernel<<<R * 128 / 256, 256, 0, stream>>>(e1o, 128, 0, e1o, 128, 0, mb, rb, ebn1g, ebn1b, R * 128, 128);
  cvt_hilo_kernel<<<R * 128 / 256, 256, 0, stream>>>(e1o, slab, slab + R * 128, R * 128);
  gemm_bf16x3_kernel<<<dim3(4, R / 64), 128, 0, stream>>>(slab, slab + R * 128, 128,
                                                          e2wbf, e2wbf + 128 * 128, e2b,
                                                          e2o, 128, 0, 128);
  col_stats_kernel<<<128, 256, 0, stream>>>(e2o, 128, 0, R, mb, rb);
  bn_relu_kernel<<<R * 128 / 256, 256, 0, stream>>>(e2o, 128, 0, cg, 640, 0, mb, rb, ebn2g, ebn2b, R * 128, 128);

  // global branch + broadcast into cg cols 128..639
  gin_max_kernel<<<8, 128, 0, stream>>>(cg, gin);
  head_kernel<<<1, 512, 0, stream>>>(gin, g1w, g1b, gbn1g, gbn1b, g2w, g2b, gbn2g, gbn2b, glob);
  bcast_glob_kernel<<<R * 512 / 256, 256, 0, stream>>>(glob, cg);

  // final projection 640 -> 512, BN+ReLU + 0.1*category bias
  cvt_hilo_kernel<<<R * 640 / 256, 256, 0, stream>>>(cg, slab, slab + R * 640, R * 640);
  gemm_bf16x3_kernel<<<dim3(16, R / 64), 128, 0, stream>>>(slab, slab + R * 640, 640,
                                                           pwbf, pwbf + 512 * 640, pb,
                                                           po, 512, 0, 640);
  col_stats_kernel<<<512, 256, 0, stream>>>(po, 512, 0, R, mb, rb);
  final_kernel<<<R * 512 / 256, 256, 0, stream>>>(po, mb, rb, pbng, pbnb, catb, cats, out);
}
